// AxialSelfAttention2d_88106959110284
// MI455X (gfx1250) — matrix operand order repaired and checked by reading, not compiled, not run on hardware
//
#include <hip/hip_runtime.h>
#include <hip/hip_bf16.h>
#include <stdint.h>

// Problem constants (B=1)
#define HD      12
#define DH      64
#define DMODEL  768
#define D3      2304
#define EE      128
#define LL      256
#define MROWS   (EE * LL)   // 32768 token rows

typedef __attribute__((ext_vector_type(16))) __bf16 v16bf;
typedef __attribute__((ext_vector_type(8)))  float  v8f;
typedef __attribute__((ext_vector_type(4)))  unsigned int v4u;
typedef __attribute__((ext_vector_type(8)))  int    v8i;
typedef __attribute__((ext_vector_type(4)))  int    v4i;

#if __has_builtin(__builtin_amdgcn_tensor_load_to_lds)
#define HAVE_TDM 1
#else
#define HAVE_TDM 0
#endif

// ---------------------------------------------------------------- helpers ---
__device__ __forceinline__ unsigned short f2bf(float f) {
  union { float f; unsigned int u; } c; c.f = f;
  unsigned int r = c.u + 0x7FFFu + ((c.u >> 16) & 1u);   // round-nearest-even
  return (unsigned short)(r >> 16);
}

// A-fragment style load: two contiguous 16-byte chunks of 8 bf16 each.
__device__ __forceinline__ v16bf load_frag_2x16B(const unsigned short* p0,
                                                 const unsigned short* p1) {
  union { v16bf v; uint4 q[2]; } u;
  u.q[0] = *(const uint4*)p0;
  u.q[1] = *(const uint4*)p1;
  return u.v;
}
// B-fragment style load: 16 contiguous bf16 (32 bytes).
__device__ __forceinline__ v16bf load_frag_32B(const unsigned short* p) {
  union { v16bf v; uint4 q[2]; } u;
  const uint4* p4 = (const uint4*)p;
  u.q[0] = p4[0];
  u.q[1] = p4[1];
  return u.v;
}

// ------------------------------------------------------------ conversions ---
__global__ void k_f32_to_bf16(const float* __restrict__ src,
                              unsigned short* __restrict__ dst, int n) {
  int i = blockIdx.x * blockDim.x + threadIdx.x;
  if (i < n) dst[i] = f2bf(src[i]);
}

__global__ void k_mask_to_neg(const int* __restrict__ mask,
                              float* __restrict__ neg, int n) {
  int i = blockIdx.x * blockDim.x + threadIdx.x;
  if (i < n) neg[i] = mask[i] ? -10000.0f : 0.0f;
}

// --------------------------------------------------------------- QKV GEMM ---
// out[m, n] = sum_k xb[m,k] * wb[n,k] + bias[n]    (M=32768, N=2304, K=768)
// grid = (M/16, N/128), block = 256 threads (8 waves); wave w owns the
// 16x16 tile at columns n0 = blockIdx.y*128 + w*16.
// Weight slab (128 x 32 bf16 per K-step) is staged into LDS by the Tensor
// Data Mover (double-buffered, TENSORcnt-synchronized); activations stream
// straight from global into A fragments.
#define KSTEPS (DMODEL / 32)   // 24

__global__ __launch_bounds__(256) void k_gemm_qkv(
    const unsigned short* __restrict__ xb,
    const unsigned short* __restrict__ wb,
    const float* __restrict__ bias,
    unsigned short* __restrict__ out) {
  __shared__ __align__(16) unsigned short bufB[2][128 * 32];   // 2 x 8KB

  const int lane    = threadIdx.x & 31;
  const int wid     = threadIdx.x >> 5;
  const int lane15  = lane & 15;
  const int halfsel = lane >> 4;           // wave32 half-group select
  const int m0   = blockIdx.x * 16;
  const int nblk = blockIdx.y * 128;
  const int n0   = nblk + wid * 16;

  // A: 16x32 bf16, lane = row, K contiguous per half-wave (k = 16*half + s)
  const unsigned short* aptr = xb + (size_t)(m0 + lane15) * DMODEL + halfsel * 16;
  // Source slab base for this block's 128 weight rows.
  const unsigned short* wslab = wb + (size_t)nblk * DMODEL;

#if HAVE_TDM
  // Issue a TDM 2-D tile load: 32 (contig K) x 128 (rows, pitch 768) bf16
  // from global into bufB[nb].  Wave 0 only (TDM ignores EXEC; per-wave op).
  auto stage = [&](int kstep, int nb) {
    if (wid != 0) return;
    unsigned lds_off = (unsigned)(uintptr_t)&bufB[nb][0];
    unsigned long long ga =
        (unsigned long long)(uintptr_t)(wslab + kstep * 32);
    // D# group 0: count=1 | lds_addr | global_addr[56:0] | type=2
    v4u g0 = { 1u, lds_off, (unsigned)ga,
               (unsigned)((ga >> 32) & 0x01FFFFFFull) | 0x80000000u };
    // D# group 1: data_size=2B; tensor 768 x 128; tile 32 x 128; stride0=768
    v8i g1 = { (int)(1u << 16),                 // wg_mask=0, data_size=1(2B)
               (int)((unsigned)DMODEL << 16),   // tensor_dim0[15:0]=768
               (int)(128u << 16),               // tensor_dim0[31:16]=0, dim1=128
               (int)(32u << 16),                // dim1[31:16]=0, tile_dim0=32
               (int)(128u),                     // tile_dim1=128, tile_dim2=0
               (int)(DMODEL),                   // tensor_dim0_stride=768
               0, 0 };
    v4i z4 = { 0, 0, 0, 0 };
#if defined(__clang_major__) && (__clang_major__ >= 23)
    v8i z8 = { 0, 0, 0, 0, 0, 0, 0, 0 };
    __builtin_amdgcn_tensor_load_to_lds(g0, g1, z4, z4, z8, 0);
#else
    __builtin_amdgcn_tensor_load_to_lds(g0, g1, z4, z4, 0);
#endif
  };
  auto wait_stage = [&]() {
    if (wid == 0) __builtin_amdgcn_s_wait_tensorcnt((short)0);
    __syncthreads();
  };
#else
  // Fallback: cooperative global->LDS copy (b128 per thread-chunk).
  auto stage = [&](int kstep, int nb) {
    for (int idx = threadIdx.x; idx < 512; idx += blockDim.x) {
      int r = idx >> 2;
      int c = (idx & 3) * 8;
      ((uint4*)&bufB[nb][0])[idx] =
          *(const uint4*)(wslab + (size_t)r * DMODEL + kstep * 32 + c);
    }
  };
  auto wait_stage = [&]() { __syncthreads(); };
#endif

  stage(0, 0);
  v8f acc = {};
  for (int i = 0; i < KSTEPS; ++i) {
    const int k0 = i * 32;
    wait_stage();                       // bufB[i&1] ready; compute(i-1) retired
    if (i + 1 < KSTEPS) stage(i + 1, (i + 1) & 1);
    __builtin_prefetch(aptr + k0 + 128, 0, 0);
    v16bf a = load_frag_2x16B(aptr + k0, aptr + k0 + 8);
    v16bf b = load_frag_32B(&bufB[i & 1][(wid * 16 + lane15) * 32 + halfsel * 16]);
    acc = __builtin_amdgcn_wmma_f32_16x16x32_bf16(false, a, false, b,
                                                  (short)0, acc, false, false);
  }

  const float bv = bias[n0 + lane15];
  const int rbase = m0 + halfsel * 8;      // C/D layout: vgpr v -> row v (+8)
#pragma unroll
  for (int v = 0; v < 8; ++v)
    out[(size_t)(rbase + v) * D3 + n0 + lane15] = f2bf(acc[v] + bv);
}

// --------------------------------------------------------- flash attention ---
// One block per (head h, outer index). Query/key token i maps to memory row
//   m(i) = outer*m_base_mul + i*m_stride        (qkv pitch = D3 elements)
// row mode: m_base_mul=L, m_stride=1, nq=nk=L(=256)
// col mode: m_base_mul=1, m_stride=L, nq=nk=E(=128)
// Mask: additive neg[m(j)] on key j. softmax over keys, O = P V.
__global__ __launch_bounds__(512) void k_axial_attn(
    const unsigned short* __restrict__ qkv,
    const float* __restrict__ neg,
    float* __restrict__ attn_out,
    int m_base_mul, int m_stride, int nq, int nk) {
  __shared__ __align__(16) unsigned short VT[64 * 272];       // V transposed [d][j]
  __shared__ __align__(16) unsigned short PB[16][16 * 40];    // per-wave P scratch

  const int h      = blockIdx.x;
  const int outer  = blockIdx.y;
  const int m_base = outer * m_base_mul;
  const int lane   = threadIdx.x & 31;
  const int wid    = threadIdx.x >> 5;
  const int nwaves = blockDim.x >> 5;
  const int lane15  = lane & 15;
  const int halfsel = lane >> 4;

  const unsigned short* qbase = qkv + h * DH;
  const unsigned short* kbase = qkv + DMODEL + h * DH;
  const unsigned short* vbase = qkv + 2 * DMODEL + h * DH;

  // Stage V (nk x 64) transposed into LDS: coalesced b128 global reads,
  // scattered 16-bit LDS writes.  Pitch 272 keeps b128 LDS reads aligned.
  for (int idx = threadIdx.x; idx < nk * 8; idx += blockDim.x) {
    int j  = idx >> 3;
    int d0 = (idx & 7) * 8;
    const unsigned short* vp =
        vbase + (size_t)(m_base + j * m_stride) * D3 + d0;
    uint4 raw = *(const uint4*)vp;
    unsigned short hh[8];
    __builtin_memcpy(hh, &raw, 16);
#pragma unroll
    for (int t = 0; t < 8; ++t) VT[(d0 + t) * 272 + j] = hh[t];
  }
  __syncthreads();

  const int akoff = halfsel * 16;

  for (int i0 = wid * 16; i0 < nq; i0 += nwaves * 16) {
    // Q tile (16x64) as two A fragments, kept in registers.
    const unsigned short* qrow =
        qbase + (size_t)(m_base + (i0 + lane15) * m_stride) * D3;
    v16bf aq0 = load_frag_2x16B(qrow + akoff,      qrow + akoff + 8);
    v16bf aq1 = load_frag_2x16B(qrow + 32 + akoff, qrow + 32 + akoff + 8);

    float mr[8], lr[8];
    v8f o0 = {}, o1 = {}, o2 = {}, o3 = {};
#pragma unroll
    for (int v = 0; v < 8; ++v) { mr[v] = -1e30f; lr[v] = 0.0f; }

    for (int j0 = 0; j0 < nk; j0 += 32) {
      // ---- S = (Q K^T) * 0.125 + neg, for a 16x32 key block ----
      const unsigned short* krow0 =
          kbase + (size_t)(m_base + (j0 + lane15) * m_stride) * D3 + halfsel * 16;
      const unsigned short* krow1 =
          kbase + (size_t)(m_base + (j0 + 16 + lane15) * m_stride) * D3 + halfsel * 16;
      v16bf bk00 = load_frag_32B(krow0);
      v16bf bk01 = load_frag_32B(krow0 + 32);
      v16bf bk10 = load_frag_32B(krow1);
      v16bf bk11 = load_frag_32B(krow1 + 32);
      v8f s0 = {}, s1 = {};
      s0 = __builtin_amdgcn_wmma_f32_16x16x32_bf16(false, aq0, false, bk00,
                                                   (short)0, s0, false, false);
      s0 = __builtin_amdgcn_wmma_f32_16x16x32_bf16(false, aq1, false, bk01,
                                                   (short)0, s0, false, false);
      s1 = __builtin_amdgcn_wmma_f32_16x16x32_bf16(false, aq0, false, bk10,
                                                   (short)0, s1, false, false);
      s1 = __builtin_amdgcn_wmma_f32_16x16x32_bf16(false, aq1, false, bk11,
                                                   (short)0, s1, false, false);

      const float n0v = neg[m_base + (j0 + lane15) * m_stride];
      const float n1v = neg[m_base + (j0 + 16 + lane15) * m_stride];

      // Row-wise max across the 16 lanes of each half-group (row lives on a
      // fixed accumulator index across lanes in the C/D layout).
      float rowm[8];
#pragma unroll
      for (int v = 0; v < 8; ++v) {
        s0[v] = s0[v] * 0.125f + n0v;
        s1[v] = s1[v] * 0.125f + n1v;
        float t = fmaxf(s0[v], s1[v]);
        t = fmaxf(t, __shfl_xor(t, 1, 32));
        t = fmaxf(t, __shfl_xor(t, 2, 32));
        t = fmaxf(t, __shfl_xor(t, 4, 32));
        t = fmaxf(t, __shfl_xor(t, 8, 32));
        rowm[v] = t;
      }

      unsigned short* pw = &PB[wid][0];
#pragma unroll
      for (int v = 0; v < 8; ++v) {
        float mnew  = fmaxf(mr[v], rowm[v]);
        float scale = __expf(mr[v] - mnew);
        mr[v] = mnew;
        float p0 = __expf(s0[v] - mnew);
        float p1 = __expf(s1[v] - mnew);
        float rs = p0 + p1;
        rs += __shfl_xor(rs, 1, 32);
        rs += __shfl_xor(rs, 2, 32);
        rs += __shfl_xor(rs, 4, 32);
        rs += __shfl_xor(rs, 8, 32);
        lr[v] = lr[v] * scale + rs;
        o0[v] *= scale; o1[v] *= scale; o2[v] *= scale; o3[v] *= scale;
        // Stage P (C-layout -> LDS) for re-shaping into A-layout.
        int r = v + halfsel * 8;
        pw[r * 40 + lane15]      = f2bf(p0);
        pw[r * 40 + 16 + lane15] = f2bf(p1);
      }

      // Re-read P as a 16x32 bf16 A fragment (wave-private LDS, in-order DS).
      v16bf pa = load_frag_2x16B(pw + lane15 * 40 + akoff,
                                 pw + lane15 * 40 + 8 + akoff);

      // ---- O += P V : B fragments are contiguous b128 reads from VT ----
      const unsigned short* vtp = &VT[lane15 * 272 + j0 + halfsel * 16];
      o0 = __builtin_amdgcn_wmma_f32_16x16x32_bf16(false, pa, false,
             load_frag_32B(vtp),            (short)0, o0, false, false);
      o1 = __builtin_amdgcn_wmma_f32_16x16x32_bf16(false, pa, false,
             load_frag_32B(vtp + 16 * 272), (short)0, o1, false, false);
      o2 = __builtin_amdgcn_wmma_f32_16x16x32_bf16(false, pa, false,
             load_frag_32B(vtp + 32 * 272), (short)0, o2, false, false);
      o3 = __builtin_amdgcn_wmma_f32_16x16x32_bf16(false, pa, false,
             load_frag_32B(vtp + 48 * 272), (short)0, o3, false, false);
    }

#pragma unroll
    for (int v = 0; v < 8; ++v) {
      float inv = 1.0f / lr[v];
      int row = v + halfsel * 8;
      size_t gm = (size_t)(m_base + (i0 + row) * m_stride);
      float* op = attn_out + gm * DMODEL + h * DH + lane15;
      op[0]  = o0[v] * inv;
      op[16] = o1[v] * inv;
      op[32] = o2[v] * inv;
      op[48] = o3[v] * inv;
    }
  }
}

// -------------------------------------------------- residual + layer norm ---
// One wave per row of 768; writes f32 result and (optionally) bf16 copy.
__global__ __launch_bounds__(256) void k_add_ln(
    const float* __restrict__ resid, const float* __restrict__ attn,
    const float* __restrict__ g, const float* __restrict__ b,
    float* __restrict__ outf, unsigned short* __restrict__ outb) {
  const int lane = threadIdx.x & 31;
  const int wid  = threadIdx.x >> 5;
  const int row  = blockIdx.x * (blockDim.x >> 5) + wid;
  const float* rp = resid + (size_t)row * DMODEL;
  const float* ap = attn + (size_t)row * DMODEL;

  float vals[24];
  float s = 0.f, sq = 0.f;
#pragma unroll
  for (int k = 0; k < 24; ++k) {
    float v = rp[lane + k * 32] + ap[lane + k * 32];
    vals[k] = v; s += v; sq += v * v;
  }
#pragma unroll
  for (int m = 1; m < 32; m <<= 1) {
    s  += __shfl_xor(s, m, 32);
    sq += __shfl_xor(sq, m, 32);
  }
  float mean = s * (1.0f / 768.0f);
  float var  = sq * (1.0f / 768.0f) - mean * mean;
  float rstd = rsqrtf(var + 1e-5f);

  float* op = outf + (size_t)row * DMODEL;
#pragma unroll
  for (int k = 0; k < 24; ++k) {
    int c = lane + k * 32;
    float o = (vals[k] - mean) * rstd * g[c] + b[c];
    op[c] = o;
    if (outb) outb[(size_t)row * DMODEL + c] = f2bf(o);
  }
}

// ------------------------------------------------------------------ driver ---
extern "C" void kernel_launch(void* const* d_in, const int* in_sizes, int n_in,
                              void* d_out, int out_size, void* d_ws, size_t ws_size,
                              hipStream_t stream) {
  const float* x     = (const float*)d_in[0];
  const float* w_row = (const float*)d_in[1];
  const float* b_row = (const float*)d_in[2];
  const float* w_col = (const float*)d_in[3];
  const float* b_col = (const float*)d_in[4];
  const float* g1    = (const float*)d_in[5];
  const float* be1   = (const float*)d_in[6];
  const float* g2    = (const float*)d_in[7];
  const float* be2   = (const float*)d_in[8];
  const int*   pmask = (const int*)d_in[9];
  float* outp = (float*)d_out;
  (void)in_sizes; (void)n_in; (void)out_size; (void)ws_size;

  char* ws = (char*)d_ws;
  size_t off = 0;
  auto walloc = [&](size_t bytes) -> char* {
    char* p = ws + off;
    off = (off + bytes + 255) & ~(size_t)255;
    return p;
  };
  unsigned short* xb   = (unsigned short*)walloc((size_t)MROWS * DMODEL * 2);
  unsigned short* wrb  = (unsigned short*)walloc((size_t)D3 * DMODEL * 2);
  unsigned short* wcb  = (unsigned short*)walloc((size_t)D3 * DMODEL * 2);
  unsigned short* qkvb = (unsigned short*)walloc((size_t)MROWS * D3 * 2);
  float* attn          = (float*)walloc((size_t)MROWS * DMODEL * 4);
  float* ln1           = (float*)walloc((size_t)MROWS * DMODEL * 4);
  float* negp          = (float*)walloc((size_t)MROWS * 4);

  { int n = MROWS * DMODEL; k_f32_to_bf16<<<(n + 255) / 256, 256, 0, stream>>>(x, xb, n); }
  { int n = D3 * DMODEL;    k_f32_to_bf16<<<(n + 255) / 256, 256, 0, stream>>>(w_row, wrb, n); }
  { int n = D3 * DMODEL;    k_f32_to_bf16<<<(n + 255) / 256, 256, 0, stream>>>(w_col, wcb, n); }
  { int n = MROWS;          k_mask_to_neg<<<(n + 255) / 256, 256, 0, stream>>>(pmask, negp, n); }

  dim3 ggrid(MROWS / 16, D3 / 128);

  // ---- row attention over L ----
  k_gemm_qkv<<<ggrid, 256, 0, stream>>>(xb, wrb, b_row, qkvb);
  k_axial_attn<<<dim3(HD, EE), 512, 0, stream>>>(qkvb, negp, attn,
                                                 /*m_base_mul=*/LL, /*m_stride=*/1,
                                                 /*nq=*/LL, /*nk=*/LL);
  k_add_ln<<<MROWS / 8, 256, 0, stream>>>(x, attn, g1, be1, ln1, xb);

  // ---- column attention over E ----
  k_gemm_qkv<<<ggrid, 256, 0, stream>>>(xb, wcb, b_col, qkvb);
  k_axial_attn<<<dim3(HD, LL), 256, 0, stream>>>(qkvb, negp, attn,
                                                 /*m_base_mul=*/1, /*m_stride=*/LL,
                                                 /*nq=*/EE, /*nk=*/EE);
  k_add_ln<<<MROWS / 8, 256, 0, stream>>>(ln1, attn, g2, be2, outp,
                                          (unsigned short*)nullptr);
}
